// CrossAttentionHead_80427557585253
// MI455X (gfx1250) — matrix operand order repaired and checked by reading, not compiled, not run on hardware
//
#include <hip/hip_runtime.h>

// Cross-attention head, fused flash-attention for MI455X (gfx1250, wave32).
// B=4, T=S=4096, C=1024, HS=64.
#define B_  4
#define T_  4096
#define S_  4096
#define C_  1024
#define HS_ 64

typedef __attribute__((ext_vector_type(16))) _Float16 v16h;
typedef __attribute__((ext_vector_type(8)))  float    v8f;
typedef __attribute__((ext_vector_type(8)))  _Float16 h8;
typedef __attribute__((ext_vector_type(4)))  float    f4;
typedef int v4i __attribute__((vector_size(4 * sizeof(int))));

#define AS1 __attribute__((address_space(1)))
#define AS3 __attribute__((address_space(3)))

// CDNA5 async global->LDS copy (ASYNCcnt-tracked), with sync fallback.
#if defined(__has_builtin)
#  if __has_builtin(__builtin_amdgcn_global_load_async_to_lds_b128) && \
      __has_builtin(__builtin_amdgcn_s_wait_asynccnt)
#    define USE_ASYNC_COPY 1
#  endif
#endif
#ifndef USE_ASYNC_COPY
#  define USE_ASYNC_COPY 0
#endif

__device__ __forceinline__ void cp_b128(void* lds, const void* gptr) {
#if USE_ASYNC_COPY
  __builtin_amdgcn_global_load_async_to_lds_b128((AS1 v4i*)gptr,
                                                 (AS3 v4i*)lds, 0, 0);
#else
  *(h8*)lds = *(const h8*)gptr;
#endif
}
__device__ __forceinline__ void cp_commit() {
#if USE_ASYNC_COPY
  __builtin_amdgcn_s_wait_asynccnt(0);
#endif
}

__device__ __forceinline__ v8f wmma_f16(v16h a, v16h b, v8f c) {
  // D = A(16x32 f16) * B(32x16 f16) + C(16x16 f32)
  return __builtin_amdgcn_wmma_f32_16x16x32_f16(
      false, a, false, b, (short)0, c, false, false);
}

// B-fragment: halves at [0..7] and [16..23] relative to p (each 16B-aligned).
__device__ __forceinline__ v16h ld_fragB(const _Float16* p) {
  h8 lo = *(const h8*)p;
  h8 hi = *(const h8*)(p + 16);
  return __builtin_shufflevector(lo, hi, 0,1,2,3,4,5,6,7,8,9,10,11,12,13,14,15);
}
// A-fragment: halves at [0..7] and [16..23] relative to p (each 16B-aligned).
__device__ __forceinline__ v16h ld_fragA(const _Float16* p) {
  h8 lo = *(const h8*)p;
  h8 hi = *(const h8*)(p + 16);
  return __builtin_shufflevector(lo, hi, 0,1,2,3,4,5,6,7,8,9,10,11,12,13,14,15);
}

// ---------------------------------------------------------------------------
// Kernel 0: Wt[64][1024] f16 = transpose(W[1024][64] f32). Tiny one-shot.
// ---------------------------------------------------------------------------
__global__ __launch_bounds__(256) void wt_kernel(const float* __restrict__ W,
                                                 _Float16* __restrict__ wT) {
  int idx = blockIdx.x * 256 + threadIdx.x;     // 65536 elements
  int k = idx >> 6, n = idx & 63;
  wT[(long)n * C_ + k] = (_Float16)W[idx];
}

// ---------------------------------------------------------------------------
// Kernel 1: out = f16( in[rows,1024] @ W[1024,64] ), W given pre-transposed.
// 4 waves/block, 16 output rows per wave. Wt K-slab in LDS [n][k], pad 40
// halfs (20-bank stride -> conflict-free b128 column reads).
// TR=true writes the output transposed per batch: out[b][hs][s] (for V).
// ---------------------------------------------------------------------------
template <bool TR>
__global__ __launch_bounds__(128) void qkv_proj_kernel(
    const float* __restrict__ in, const _Float16* __restrict__ wT,
    _Float16* __restrict__ out)
{
  __shared__ __align__(16) _Float16 w_lds[64 * 40];

  const int tid  = threadIdx.x;
  const int wave = tid >> 5;
  const int lane = tid & 31;
  const int m    = lane & 15;
  const int kbA  = (lane < 16) ? 0 : 8;
  const int n    = lane & 15;
  const int kbB  = (lane < 16) ? 0 : 8;
  const long row0 = (long)blockIdx.x * 64 + wave * 16;

  v8f zero = {};
  v8f acc[4] = {zero, zero, zero, zero};

  for (int kc = 0; kc < C_ / 32; ++kc) {
    __syncthreads();
#pragma unroll
    for (int it = 0; it < 2; ++it) {
      int idx = tid + it * 128;                 // 256 h8 groups
      int nn  = idx >> 2;                       // 0..63
      int ko  = (idx & 3) << 3;                 // 0,8,16,24
      cp_b128(w_lds + nn * 40 + ko, wT + (long)nn * C_ + kc * 32 + ko);
    }
    cp_commit();
    __syncthreads();

    // A fragment: 4x global_load_b128 of f32 + pack to f16.
    const float* arow = in + (row0 + m) * C_ + kc * 32;
    f4 x0 = *(const f4*)(arow + kbA);
    f4 x1 = *(const f4*)(arow + kbA + 4);
    f4 x2 = *(const f4*)(arow + kbA + 16);
    f4 x3 = *(const f4*)(arow + kbA + 20);
    v16h a;
#pragma unroll
    for (int j = 0; j < 4; ++j) {
      a[j]      = (_Float16)x0[j];
      a[4 + j]  = (_Float16)x1[j];
      a[8 + j]  = (_Float16)x2[j];
      a[12 + j] = (_Float16)x3[j];
    }
    // Load all 4 B-fragments first, then burst the WMMAs (amortize dscnt).
    v16h bf[4];
#pragma unroll
    for (int t = 0; t < 4; ++t)
      bf[t] = ld_fragB(w_lds + (t * 16 + n) * 40 + kbB);
#pragma unroll
    for (int t = 0; t < 4; ++t)
      acc[t] = wmma_f16(a, bf[t], acc[t]);
  }

#pragma unroll
  for (int t = 0; t < 4; ++t)
#pragma unroll
    for (int r = 0; r < 8; ++r) {
      int rowm = (lane < 16) ? r : (r + 8);
      long gr  = row0 + rowm;
      int col  = t * 16 + n;
      long idx;
      if (TR) {                                 // out[b][hs][s]
        long bb = gr >> 12;                     // S_ == 4096
        long s  = gr & (S_ - 1);
        idx = (bb * HS_ + col) * S_ + s;
      } else {                                  // out[row][hs]
        idx = gr * HS_ + col;
      }
      out[idx] = (_Float16)acc[t][r];
    }
}

// ---------------------------------------------------------------------------
// Kernel 2: fused flash attention. 4 waves/block = 64 query rows, 256 blocks.
// Double-buffered K [key][hs] and V^T [hs][key] tiles in LDS (pad 72 halfs
// = 36-bank stride -> conflict-free b128 reads). Next tile's copy is issued
// (async where available) before computing the current one.
// ---------------------------------------------------------------------------
__device__ __forceinline__ void stage_tile(_Float16* kbuf, _Float16* vbuf,
                                           const _Float16* kp,
                                           const _Float16* vp, int tid) {
#pragma unroll
  for (int it = 0; it < 4; ++it) {
    int idx = tid + it * 128;                   // 512 h8 groups
    int r   = idx >> 3;                         // 0..63
    int o8  = (idx & 7) << 3;                   // 0..56
    cp_b128(kbuf + r * 72 + o8, kp + (long)r * HS_ + o8);
    cp_b128(vbuf + r * 72 + o8, vp + (long)r * S_  + o8);
  }
}

__global__ __launch_bounds__(128) void flash_attn_kernel(
    const _Float16* __restrict__ qh, const _Float16* __restrict__ kh,
    const _Float16* __restrict__ vt, float* __restrict__ out)
{
  __shared__ __align__(16) _Float16 k_lds [2][64 * 72];
  __shared__ __align__(16) _Float16 vT_lds[2][64 * 72];
  __shared__ __align__(16) _Float16 p_lds [4 * 16 * 72];

  const int tid  = threadIdx.x;
  const int wave = tid >> 5;
  const int lane = tid & 31;
  const int m    = lane & 15;
  const int kbA  = (lane < 16) ? 0 : 8;
  const int n    = lane & 15;
  const int kbB  = (lane < 16) ? 0 : 8;

  const int  tBlocks = T_ / 64;
  const int  b     = blockIdx.x / tBlocks;
  const int  tb    = blockIdx.x % tBlocks;
  const long qbase = (long)b * T_ + tb * 64 + wave * 16;

  const _Float16* kbase = kh + (long)b * S_ * HS_;   // [key][hs]
  const _Float16* vbase = vt + (long)b * HS_ * S_;   // [hs][key]

  const float SCL = 0.125f * 1.44269504088896f;      // 1/sqrt(64) * log2(e)

  // Q fragments, pre-scaled by SCL (folds softmax scale + log2e into QK^T).
  v16h aq[2];
#pragma unroll
  for (int c = 0; c < 2; ++c) {
    aq[c] = ld_fragA(qh + (qbase + m) * HS_ + c * 32 + kbA);
#pragma unroll
    for (int i = 0; i < 16; ++i) aq[c][i] = aq[c][i] * (_Float16)SCL;
  }

  v8f zero = {};
  v8f o[4] = {zero, zero, zero, zero};
  float m_i[8], l_i[8];
#pragma unroll
  for (int r = 0; r < 8; ++r) { m_i[r] = -3.0e38f; l_i[r] = 0.0f; }

  _Float16* pw = p_lds + wave * 16 * 72;

  // Prologue: stage tile 0 into buffer 0.
  stage_tile(k_lds[0], vT_lds[0], kbase, vbase, tid);
  cp_commit();
  __syncthreads();

  for (int s0 = 0; s0 < S_; s0 += 64) {
    const int cur = (s0 >> 6) & 1;
    // Issue next tile's copy into the other buffer (overlaps with compute).
    if (s0 + 64 < S_)
      stage_tile(k_lds[cur ^ 1], vT_lds[cur ^ 1],
                 kbase + (long)(s0 + 64) * HS_, vbase + s0 + 64, tid);
    if (s0 + 128 < S_) {                         // prefetch tile+2 into L2
      __builtin_prefetch(kbase + (long)(s0 + 128) * HS_ + tid * 32, 0, 0);
      __builtin_prefetch(vbase + s0 + 128 + (long)(tid & 63) * S_, 0, 0);
    }

    // ---- scores: 16x64 = Qs(16x64) * K_tile^T (pre-scaled) ----
    v8f sc[4] = {zero, zero, zero, zero};
#pragma unroll
    for (int c = 0; c < 2; ++c) {
      v16h bk[4];
#pragma unroll
      for (int t = 0; t < 4; ++t)
        bk[t] = ld_fragB(&k_lds[cur][(t * 16 + n) * 72 + c * 32 + kbB]);
#pragma unroll
      for (int t = 0; t < 4; ++t)
        sc[t] = wmma_f16(aq[c], bk[t], sc[t]);
    }

    // ---- online softmax, base-2 domain (rows live across 16-lane halves) ----
#pragma unroll
    for (int r = 0; r < 8; ++r) {
      float v = fmaxf(fmaxf(sc[0][r], sc[1][r]), fmaxf(sc[2][r], sc[3][r]));
#pragma unroll
      for (int off = 1; off < 16; off <<= 1)
        v = fmaxf(v, __shfl_xor(v, off, 16));
      float mn   = fmaxf(m_i[r], v);
      float corr = exp2f(m_i[r] - mn);
      m_i[r] = mn;
      l_i[r] *= corr;
#pragma unroll
      for (int t = 0; t < 4; ++t) o[t][r] *= corr;
      float rs = 0.0f;
#pragma unroll
      for (int t = 0; t < 4; ++t) {
        float p = exp2f(sc[t][r] - mn);
        sc[t][r] = p;
        rs += p;
      }
#pragma unroll
      for (int off = 1; off < 16; off <<= 1)
        rs += __shfl_xor(rs, off, 16);
      l_i[r] += rs;
    }

    // ---- P: C/D layout -> per-wave LDS (same-wave LDS ops are in-order) ----
#pragma unroll
    for (int t = 0; t < 4; ++t)
#pragma unroll
      for (int r = 0; r < 8; ++r) {
        int rowm = (lane < 16) ? r : (r + 8);
        pw[rowm * 72 + t * 16 + n] = (_Float16)sc[t][r];
      }

    // ---- O += P(16x64) * V_tile, V^T in LDS -> contiguous B-frags ----
#pragma unroll
    for (int c = 0; c < 2; ++c) {
      v16h ap = ld_fragA(pw + m * 72 + c * 32 + kbA);
      v16h bv[4];
#pragma unroll
      for (int t = 0; t < 4; ++t)
        bv[t] = ld_fragB(&vT_lds[cur][(t * 16 + n) * 72 + c * 32 + kbB]);
#pragma unroll
      for (int t = 0; t < 4; ++t)
        o[t] = wmma_f16(ap, bv[t], o[t]);
    }

    cp_commit();          // next tile's async copies have landed
    __syncthreads();      // all waves done with cur before it is re-staged
  }

  // ---- epilogue: O / l ----
#pragma unroll
  for (int r = 0; r < 8; ++r) {
    float inv  = 1.0f / l_i[r];
    int   rowm = (lane < 16) ? r : (r + 8);
#pragma unroll
    for (int t = 0; t < 4; ++t)
      out[(qbase + rowm) * HS_ + t * 16 + n] = o[t][r] * inv;
  }
}

// ---------------------------------------------------------------------------
extern "C" void kernel_launch(void* const* d_in, const int* in_sizes, int n_in,
                              void* d_out, int out_size, void* d_ws, size_t ws_size,
                              hipStream_t stream) {
  const float* x  = (const float*)d_in[0];
  const float* eo = (const float*)d_in[1];
  const float* Wq = (const float*)d_in[2];
  const float* Wk = (const float*)d_in[3];
  const float* Wv = (const float*)d_in[4];
  float* out = (float*)d_out;

  // Workspace layout (f16): q[16384,64] | k[16384,64] | vT[4][64][4096]
  //                         | WqT | WkT | WvT (each [64][1024])  ~= 6.4 MB
  _Float16* qh  = (_Float16*)d_ws;
  _Float16* kh  = qh + (long)B_ * T_ * HS_;
  _Float16* vth = kh + (long)B_ * S_ * HS_;
  _Float16* wqt = vth + (long)B_ * S_ * HS_;
  _Float16* wkt = wqt + (long)C_ * HS_;
  _Float16* wvt = wkt + (long)C_ * HS_;

  wt_kernel<<<256, 256, 0, stream>>>(Wq, wqt);
  wt_kernel<<<256, 256, 0, stream>>>(Wk, wkt);
  wt_kernel<<<256, 256, 0, stream>>>(Wv, wvt);

  dim3 blk(128);
  qkv_proj_kernel<false><<<(B_ * T_) / 64, blk, 0, stream>>>(x,  wqt, qh);
  qkv_proj_kernel<false><<<(B_ * S_) / 64, blk, 0, stream>>>(eo, wkt, kh);
  qkv_proj_kernel<true ><<<(B_ * S_) / 64, blk, 0, stream>>>(eo, wvt, vth);
  flash_attn_kernel<<<B_ * (T_ / 64), blk, 0, stream>>>(qh, kh, vth, out);
}
